// LearnedRigidBody_9311489097932
// MI455X (gfx1250) — compile-verified, matrix operand order checked by reading, not run on hardware
//
#include <hip/hip_runtime.h>

// ---------------------------------------------------------------------------
// LearnedRigidBody forward (qddot, qddot_solve) on MI455X (gfx1250, wave32).
// One wave per workgroup = 16 samples end-to-end. jacfwd == forward mode with
// 7 identity tangents -> mm/pot stages are M=128 GEMMs on v_wmma_f32_16x16x32_f16
// (one sample's primal+7 tangent rows land in a single lane's 8 accumulators,
// so the tanh'/softplus' chain rule is lane-local). gf/ud stages are M=16 GEMMs.
// M = L L^T from the Cholesky net -> solve is two triangular substitutions.
// LDS budget 32 KB/WG (sA ping-pong + f32 scratch overlaid on dead sB).
// ---------------------------------------------------------------------------

typedef __attribute__((ext_vector_type(16))) _Float16 v16h;
typedef __attribute__((ext_vector_type(8)))  float    v8f;
typedef __attribute__((ext_vector_type(4)))  float    v4f;

#define LDS_FENCE() asm volatile("s_wait_dscnt 0" ::: "memory")

struct Ptrs { const float* p[44]; };

// ---- layer tables (20 layers: mm0..4, pot0..4, gf0..4, ud0..4) ----
constexpr int kKin [20] = {7,16,32,64,64,  7,16,32,32,16,  14,32,64,64,16,  29,64,64,32,16};
constexpr int kNout[20] = {16,32,64,64,28, 16,32,32,16,1,  32,64,64,16,7,   64,64,32,16,7};
constexpr int kKpad[20] = {32,32,32,64,64, 32,32,32,32,32, 32,32,64,64,32,  32,64,64,32,32};
constexpr int kNpad[20] = {32,32,64,64,32, 32,32,32,32,32, 32,64,64,32,32,  64,64,32,32,32};
constexpr int kWoff[20] = {0,1024,2048,4096,8192, 10240,11264,12288,13312,14336,
                           15360,16384,18432,22528,24576, 25600,27648,31744,33792,34816};
constexpr int kBoff[20] = {0,32,64,128,192, 224,256,288,320,352,
                           384,416,480,544,576, 608,672,736,768,800};
constexpr int kWIdx[20] = {4,6,8,10,12, 14,16,18,20,22, 24,26,28,30,32, 34,36,38,40,42};
constexpr int WTOTAL_H = 35840;   // packed f16 weights

// ---------------------------------------------------------------------------
// Fast branch-free activations (CDNA5 TRANS pipeline; v_tanh_f32 if declared).
// ---------------------------------------------------------------------------
constexpr float kLog2e  = 1.44269504088896340736f;
constexpr float kLn2    = 0.69314718055994530942f;

__device__ __forceinline__ float fexp2(float x) { return __builtin_amdgcn_exp2f(x); }
__device__ __forceinline__ float flog2(float x) { return __builtin_amdgcn_logf(x); }
__device__ __forceinline__ float frcp(float x)  { return __builtin_amdgcn_rcpf(x); }

__device__ __forceinline__ float fast_tanh(float x) {
#if __has_builtin(__builtin_amdgcn_tanhf)
  return __builtin_amdgcn_tanhf(x);
#else
  // tanh(x) = 1 - 2/(e^{2x}+1); exp2 saturates to +inf / 0 => tanh -> +/-1.
  float e = fexp2(x * (2.f * kLog2e));
  return 1.f - 2.f * frcp(e + 1.f);
#endif
}
__device__ __forceinline__ float fast_sigmoid(float x) {
  return frcp(1.f + fexp2(-kLog2e * x));
}
__device__ __forceinline__ float fast_softplus(float x) {
  // log1p(exp(x)); for large x exp2 -> inf so select linear branchlessly.
  float sp = kLn2 * flog2(1.f + fexp2(kLog2e * x));
  return (x > 20.f) ? x : sp;
}

// ---------------------------------------------------------------------------
// Weight prep: W (Kin x Nout, row-major f32) -> Wt[n][k] f16, zero-padded to
// (Npad x Kpad) so WMMA B-fragments are two contiguous 16B loads per lane.
// ---------------------------------------------------------------------------
__global__ void lrb_pack(Ptrs ptrs, _Float16* __restrict__ wbuf, float* __restrict__ bbuf) {
  const int id = blockIdx.x;
  const float* W = ptrs.p[kWIdx[id]];
  const float* b = ptrs.p[kWIdx[id] + 1];
  const int Kin = kKin[id], Nout = kNout[id], Kpad = kKpad[id], Npad = kNpad[id];
  _Float16* wd = wbuf + kWoff[id];
  for (int i = threadIdx.x; i < Npad * Kpad; i += blockDim.x) {
    const int n = i / Kpad, k = i - n * Kpad;
    const float val = (n < Nout && k < Kin) ? W[k * Nout + n] : 0.f;
    wd[i] = (_Float16)val;
  }
  float* bd = bbuf + kBoff[id];
  for (int i = threadIdx.x; i < Npad; i += blockDim.x)
    bd[i] = (i < Nout) ? b[i] : 0.f;
}

// ---------------------------------------------------------------------------
// Fragment loaders (CDNA5 16-bit A/B layout). Activation pitch = 64 f16.
// ---------------------------------------------------------------------------
__device__ __forceinline__ v16h ldsFragA(const _Float16* buf, int m0, int kbase, int lane) {
  const int m = lane & 15, g = lane >> 4;
  const _Float16* p = buf + (m0 + m) * 64 + kbase + 8 * g;
  union { v16h h; v4f f[2]; } u;
  u.f[0] = *(const v4f*)(p);
  u.f[1] = *(const v4f*)(p + 16);
  return u.h;
}
__device__ __forceinline__ v16h ldFragB(const _Float16* wt, int Kpad, int n0, int kbase, int lane) {
  const int n = n0 + (lane & 15), g = lane >> 4;
  const _Float16* p = wt + n * Kpad + kbase + 8 * g;
  union { v16h h; v4f f[2]; } u;
  u.f[0] = *(const v4f*)(p);
  u.f[1] = *(const v4f*)(p + 16);
  return u.h;
}
__device__ __forceinline__ v8f wmma32(v16h a, v16h b, v8f c) {
  return __builtin_amdgcn_wmma_f32_16x16x32_f16(false, a, false, b, (short)0, c, false, false);
}

// ---------------------------------------------------------------------------
// Hidden layer, jacfwd stages: M=128 rows (16 samples x [primal + 7 tangents]).
// ---------------------------------------------------------------------------
__device__ __forceinline__ void layerJac(const _Float16* src, _Float16* dst,
                                         const _Float16* wt, const float* bias,
                                         int Kpad, int Npad, int lane) {
  const int ntiles = Npad >> 4;
  const bool two = (Kpad > 32);
  const int n = lane & 15, g = lane >> 4;
  for (int mt = 0; mt < 8; ++mt) {
    v16h a0 = ldsFragA(src, mt * 16, 0, lane);
    v16h a1 = {};
    if (two) a1 = ldsFragA(src, mt * 16, 32, lane);
    for (int nt = 0; nt < ntiles; ++nt) {
      v8f acc = {};
      acc = wmma32(a0, ldFragB(wt, Kpad, nt * 16, 0, lane), acc);
      if (two) acc = wmma32(a1, ldFragB(wt, Kpad, nt * 16, 32, lane), acc);
      const int col = nt * 16 + n;
      const float z0 = acc[0] + bias[col];          // primal pre-activation
      const float t  = fast_tanh(z0);
      const float s  = 1.f - t * t;                 // tanh'
      _Float16* d = dst + (mt * 16 + 8 * g) * 64 + col;
      d[0] = (_Float16)t;
#pragma unroll
      for (int r = 1; r < 8; ++r) d[r * 64] = (_Float16)(s * acc[r]);  // tangents: no bias
    }
  }
}

// Hidden layer, plain stages (M=16: each row is a sample).
__device__ __forceinline__ void layerP16(const _Float16* src, _Float16* dst,
                                         const _Float16* wt, const float* bias,
                                         int Kpad, int Npad, int lane) {
  const int ntiles = Npad >> 4;
  const bool two = (Kpad > 32);
  const int n = lane & 15, g = lane >> 4;
  v16h a0 = ldsFragA(src, 0, 0, lane);
  v16h a1 = {};
  if (two) a1 = ldsFragA(src, 0, 32, lane);
  for (int nt = 0; nt < ntiles; ++nt) {
    v8f acc = {};
    acc = wmma32(a0, ldFragB(wt, Kpad, nt * 16, 0, lane), acc);
    if (two) acc = wmma32(a1, ldFragB(wt, Kpad, nt * 16, 32, lane), acc);
    const int col = nt * 16 + n;
    const float bn = bias[col];
#pragma unroll
    for (int r = 0; r < 8; ++r)
      dst[(8 * g + r) * 64 + col] = (_Float16)fast_tanh(acc[r] + bn);
  }
}

// Final mm layer: linear 28 outputs; softplus(+eps)/sigmoid chain on the 7
// diagonal tril indices. Writes f32 [sample][row 0..7][28].
__device__ __forceinline__ void layerMMFinal(const _Float16* src, float* ldl,
                                             const _Float16* wt, const float* bias, int lane) {
  const int n = lane & 15, g = lane >> 4;
  for (int mt = 0; mt < 8; ++mt) {
    v16h a0 = ldsFragA(src, mt * 16, 0, lane);
    v16h a1 = ldsFragA(src, mt * 16, 32, lane);
    for (int nt = 0; nt < 2; ++nt) {
      v8f acc = {};
      acc = wmma32(a0, ldFragB(wt, 64, nt * 16, 0, lane), acc);
      acc = wmma32(a1, ldFragB(wt, 64, nt * 16, 32, lane), acc);
      const int col = nt * 16 + n;
      if (col < 28) {
        const bool diag = (col == 0) | (col == 2) | (col == 5) | (col == 9) |
                          (col == 14) | (col == 20) | (col == 27);
        const int s = mt * 2 + g;
        float* d = ldl + s * 224 + col;
        const float z0 = acc[0] + bias[col];
        if (diag) {
          const float sg = fast_sigmoid(z0);        // d softplus / dz
          d[0] = fast_softplus(z0) + 1e-3f;
#pragma unroll
          for (int r = 1; r < 8; ++r) d[r * 28] = sg * acc[r];
        } else {
          d[0] = z0;
#pragma unroll
          for (int r = 1; r < 8; ++r) d[r * 28] = acc[r];
        }
      }
    }
  }
}

// Final pot layer: scalar output; gradient = tangent rows 1..7 of column 0.
__device__ __forceinline__ void layerPotFinal(const _Float16* src, float* gbuf,
                                              const _Float16* wt, int lane) {
  const int n = lane & 15, g = lane >> 4;
  for (int mt = 0; mt < 8; ++mt) {
    v16h a0 = ldsFragA(src, mt * 16, 0, lane);
    v8f acc = {};
    acc = wmma32(a0, ldFragB(wt, 32, 0, 0, lane), acc);
    if (n == 0) {
      const int s = mt * 2 + g;
#pragma unroll
      for (int r = 1; r < 8; ++r) gbuf[s * 7 + (r - 1)] = acc[r];
    }
  }
}

// Final gf layer: linear, F[sample][0..6] -> f32 LDS (stride 16).
__device__ __forceinline__ void layerGFFinal(const _Float16* src, float* fbuf,
                                             const _Float16* wt, const float* bias, int lane) {
  const int n = lane & 15, g = lane >> 4;
  v16h a0 = ldsFragA(src, 0, 0, lane);
  v8f acc = {};
  acc = wmma32(a0, ldFragB(wt, 32, 0, 0, lane), acc);
  if (n < 7) {
    const float bn = bias[n];
#pragma unroll
    for (int r = 0; r < 8; ++r) fbuf[(8 * g + r) * 16 + n] = acc[r] + bn;
  }
}

// Final ud layer: qddot = mlp_out + qddot_solve, straight to global.
__device__ __forceinline__ void layerUDFinal(const _Float16* src, const float* qs,
                                             const _Float16* wt, const float* bias,
                                             float* out, int base, int lane) {
  const int n = lane & 15, g = lane >> 4;
  v16h a0 = ldsFragA(src, 0, 0, lane);
  v8f acc = {};
  acc = wmma32(a0, ldFragB(wt, 32, 0, 0, lane), acc);
  if (n < 7) {
    const float bn = bias[n];
#pragma unroll
    for (int r = 0; r < 8; ++r) {
      const int m = 8 * g + r;
      out[(base + m) * 7 + n] = acc[r] + bn + qs[m * 7 + n];
    }
  }
}

// ---- input builders ----
__device__ __forceinline__ void buildJacInput(_Float16* dst, const float* q, int base, int lane) {
  for (int idx = lane; idx < 16 * 8 * 32; idx += 32) {
    const int col = idx & 31, row = idx >> 5;
    const int s = row >> 3, r = row & 7;
    float val = 0.f;
    if (r == 0) { if (col < 7) val = q[(base + s) * 7 + col]; }
    else if (col == r - 1) val = 1.f;                 // identity tangent seeds
    dst[row * 64 + col] = (_Float16)val;
  }
}
__device__ __forceinline__ void buildGFInput(_Float16* dst, const float* q, const float* v,
                                             int base, int lane) {
  for (int idx = lane; idx < 16 * 32; idx += 32) {
    const int col = idx & 31, s = idx >> 5;
    float val = 0.f;
    if (col < 7)       val = q[(base + s) * 7 + col];
    else if (col < 14) val = v[(base + s) * 7 + (col - 7)];
    dst[s * 64 + col] = (_Float16)val;
  }
}
__device__ __forceinline__ void buildUDInput(_Float16* dst, const float* q, const float* v,
                                             const float* u, const float* dt, const float* qs,
                                             int base, int lane) {
  for (int idx = lane; idx < 16 * 32; idx += 32) {
    const int col = idx & 31, s = idx >> 5;
    float val = 0.f;
    if (col < 7)       val = q[(base + s) * 7 + col];
    else if (col < 14) val = v[(base + s) * 7 + col - 7];
    else if (col < 21) val = qs[s * 7 + col - 14];
    else if (col < 28) val = u[(base + s) * 7 + col - 21];
    else if (col == 28) val = dt[base + s];
    dst[s * 64 + col] = (_Float16)val;
  }
}

// ---------------------------------------------------------------------------
// Per-sample epilogue (lanes 0-15, sample = lane): Coriolis from dL_k, rhs,
// and M x = rhs via L / L^T triangular substitution.  tril: (r,c)->r(r+1)/2+c.
// ---------------------------------------------------------------------------
__device__ __forceinline__ void sampleEpilogue(const float* ldl, const float* gbuf,
                                               const float* fbuf, const float* v,
                                               const float* u, float* qs, float* out,
                                               int base, int outHalf, int lane) {
  if (lane >= 16) return;
  const int s = lane;
  const float* lp = ldl + s * 224;
  float l[28];
#pragma unroll
  for (int m = 0; m < 28; ++m) l[m] = lp[m];
  float vv[7];
#pragma unroll
  for (int i = 0; i < 7; ++i) vv[i] = v[(base + s) * 7 + i];

  float p[7];                                     // p = L^T v
  for (int c = 0; c < 7; ++c) {
    float a = 0.f;
    for (int r = c; r < 7; ++r) a += l[r * (r + 1) / 2 + c] * vv[r];
    p[c] = a;
  }
  float cacc[7], eacc[7], S[7];
  for (int i = 0; i < 7; ++i) { cacc[i] = 0.f; eacc[i] = 0.f; }
  for (int k = 0; k < 7; ++k) {
    const float* dl = lp + (k + 1) * 28;          // dL/dq_k (tril, diag pre-scaled)
    float ck[7], ek[7];
    for (int r = 0; r < 7; ++r) {                 // ck = dL_k p
      float a = 0.f;
      for (int c = 0; c <= r; ++c) a += dl[r * (r + 1) / 2 + c] * p[c];
      ck[r] = a;
    }
    for (int c = 0; c < 7; ++c) {                 // ek = dL_k^T v
      float a = 0.f;
      for (int r = c; r < 7; ++r) a += dl[r * (r + 1) / 2 + c] * vv[r];
      ek[c] = a;
    }
    float sk = 0.f;
    for (int i = 0; i < 7; ++i) sk += vv[i] * ck[i];
    S[k] = 2.f * sk;                              // v^T dM_k v
    for (int i = 0; i < 7; ++i) { cacc[i] += vv[k] * ck[i]; eacc[i] += vv[k] * ek[i]; }
  }
  float rhs[7];
  for (int r = 0; r < 7; ++r) {
    float Le = 0.f;
    for (int c = 0; c <= r; ++c) Le += l[r * (r + 1) / 2 + c] * eacc[c];
    const float Cv = cacc[r] + Le - 0.5f * S[r];
    rhs[r] = u[(base + s) * 7 + r] + fbuf[s * 16 + r] - Cv - gbuf[s * 7 + r];
  }
  float dinv[7];
#pragma unroll
  for (int i = 0; i < 7; ++i) dinv[i] = frcp(l[i * (i + 1) / 2 + i]);  // diag >= 1e-3
  float y[7];                                     // forward substitution L y = rhs
  for (int r = 0; r < 7; ++r) {
    float a = rhs[r];
    for (int c = 0; c < r; ++c) a -= l[r * (r + 1) / 2 + c] * y[c];
    y[r] = a * dinv[r];
  }
  float x[7];                                     // back substitution L^T x = y
  for (int r = 6; r >= 0; --r) {
    float a = y[r];
    for (int c = r + 1; c < 7; ++c) a -= l[c * (c + 1) / 2 + r] * x[c];
    x[r] = a * dinv[r];
  }
  for (int i = 0; i < 7; ++i) {
    qs[s * 7 + i] = x[i];
    out[outHalf + (base + s) * 7 + i] = x[i];     // qddot_solve
  }
}

// ---------------------------------------------------------------------------
// Main fused kernel. LDS plan (32 KB total -> ~10 waves/WGP):
//   mm stage : full-width ping-pong sA <-> sB; final writes sLDL (on sB).
//   pot stage: 32-wide -> ping-pong between column halves of sA.
//   gf/ud    : M=16   -> ping-pong between row halves of sA.
//   f32 scratch (sLDL/sG/sF/sQS, 16.3 KB) overlaid on sB once mm is done.
// ---------------------------------------------------------------------------
__global__ __launch_bounds__(32) void lrb_main(
    const float* __restrict__ q, const float* __restrict__ v,
    const float* __restrict__ u, const float* __restrict__ dt,
    const _Float16* __restrict__ wbuf, const float* __restrict__ bbuf,
    float* __restrict__ out, int outHalf) {
  __shared__ __align__(16) _Float16 sA[128 * 64];
  __shared__ __align__(16) _Float16 sB[128 * 64];
  float* sLDL = (float*)sB;              // 16*224 = 3584 floats
  float* sG   = (float*)sB + 3584;       // 112 floats
  float* sF   = (float*)sB + 3696;       // 256 floats
  float* sQS  = (float*)sB + 3952;       // 112 floats (total 16256 B <= 16384)

  _Float16* cA0 = sA;                    // column halves (pot stage)
  _Float16* cA1 = sA + 32;
  _Float16* rA0 = sA;                    // row halves (gf/ud stages)
  _Float16* rA1 = sA + 16 * 64;

  const int lane = threadIdx.x;
  const int base = blockIdx.x * 16;

  // ---- mass matrix MLP + jacfwd (M = 128 rows) ----
  buildJacInput(sA, q, base, lane);                                        LDS_FENCE();
  layerJac(sA, sB, wbuf + kWoff[0], bbuf + kBoff[0], 32, 32, lane);        LDS_FENCE();
  layerJac(sB, sA, wbuf + kWoff[1], bbuf + kBoff[1], 32, 32, lane);        LDS_FENCE();
  layerJac(sA, sB, wbuf + kWoff[2], bbuf + kBoff[2], 32, 64, lane);        LDS_FENCE();
  layerJac(sB, sA, wbuf + kWoff[3], bbuf + kBoff[3], 64, 64, lane);        LDS_FENCE();
  layerMMFinal(sA, sLDL, wbuf + kWoff[4], bbuf + kBoff[4], lane);          LDS_FENCE();

  // ---- potential gradient via forward mode (M = 128, 32-wide) ----
  buildJacInput(cA0, q, base, lane);                                       LDS_FENCE();
  layerJac(cA0, cA1, wbuf + kWoff[5], bbuf + kBoff[5], 32, 32, lane);      LDS_FENCE();
  layerJac(cA1, cA0, wbuf + kWoff[6], bbuf + kBoff[6], 32, 32, lane);      LDS_FENCE();
  layerJac(cA0, cA1, wbuf + kWoff[7], bbuf + kBoff[7], 32, 32, lane);      LDS_FENCE();
  layerJac(cA1, cA0, wbuf + kWoff[8], bbuf + kBoff[8], 32, 32, lane);      LDS_FENCE();
  layerPotFinal(cA0, sG, wbuf + kWoff[9], lane);                           LDS_FENCE();

  // ---- generalized force MLP (M = 16) ----
  buildGFInput(rA0, q, v, base, lane);                                     LDS_FENCE();
  layerP16(rA0, rA1, wbuf + kWoff[10], bbuf + kBoff[10], 32, 32, lane);    LDS_FENCE();
  layerP16(rA1, rA0, wbuf + kWoff[11], bbuf + kBoff[11], 32, 64, lane);    LDS_FENCE();
  layerP16(rA0, rA1, wbuf + kWoff[12], bbuf + kBoff[12], 64, 64, lane);    LDS_FENCE();
  layerP16(rA1, rA0, wbuf + kWoff[13], bbuf + kBoff[13], 64, 32, lane);    LDS_FENCE();
  layerGFFinal(rA0, sF, wbuf + kWoff[14], bbuf + kBoff[14], lane);         LDS_FENCE();

  // ---- Coriolis + rhs + SPD solve ----
  sampleEpilogue(sLDL, sG, sF, v, u, sQS, out, base, outHalf, lane);       LDS_FENCE();

  // ---- residual dynamics MLP (M = 16) ----
  buildUDInput(rA0, q, v, u, dt, sQS, base, lane);                         LDS_FENCE();
  layerP16(rA0, rA1, wbuf + kWoff[15], bbuf + kBoff[15], 32, 64, lane);    LDS_FENCE();
  layerP16(rA1, rA0, wbuf + kWoff[16], bbuf + kBoff[16], 64, 64, lane);    LDS_FENCE();
  layerP16(rA0, rA1, wbuf + kWoff[17], bbuf + kBoff[17], 64, 32, lane);    LDS_FENCE();
  layerP16(rA1, rA0, wbuf + kWoff[18], bbuf + kBoff[18], 32, 32, lane);    LDS_FENCE();
  layerUDFinal(rA0, sQS, wbuf + kWoff[19], bbuf + kBoff[19], out, base, lane);
}

extern "C" void kernel_launch(void* const* d_in, const int* in_sizes, int n_in,
                              void* d_out, int out_size, void* d_ws, size_t ws_size,
                              hipStream_t stream) {
  (void)ws_size;
  Ptrs ptrs;
  for (int i = 0; i < 44; ++i) ptrs.p[i] = (i < n_in) ? (const float*)d_in[i] : nullptr;

  _Float16* wbuf = (_Float16*)d_ws;
  float* bbuf = (float*)((char*)d_ws + WTOTAL_H * sizeof(_Float16));

  lrb_pack<<<20, 256, 0, stream>>>(ptrs, wbuf, bbuf);

  const int nB = in_sizes[0] / 7;        // batch size (131072)
  const int outHalf = out_size / 2;      // qddot | qddot_solve split
  lrb_main<<<nB / 16, 32, 0, stream>>>(
      (const float*)d_in[0], (const float*)d_in[1], (const float*)d_in[2],
      (const float*)d_in[3], wbuf, bbuf, (float*)d_out, outHalf);
}
